// LongformerEncoder_69020124446892
// MI455X (gfx1250) — compile-verified, matrix operand order checked by reading, not run on hardware
//
#include <hip/hip_runtime.h>
#include <hip/hip_bf16.h>

typedef __attribute__((ext_vector_type(16))) _Float16 v16h;
typedef __attribute__((ext_vector_type(8)))  _Float16 v8h;
typedef __attribute__((ext_vector_type(8)))  float    v8f;

#define S_LEN   2048
#define D_DIM   512
#define H_NUM   8
#define DH_DIM  64
#define M_DIM   2048
#define L_NUM   6
#define WIN_HALF 256
#define NEG_BIAS (-1e9f)

// LDS tile strides (halves): multiples of 8 halves = 16B so b128 loads stay
// aligned, and co-prime-ish with the 64-bank layout to spread conflicts.
#define SA_STRIDE 40   // 16-row-major tiles (A fragments, P tiles)
#define SK_STRIDE 72   // 32-row K/V/B tiles

// ---------------------------------------------------------------------------
// WMMA fragment lane mappings (CDNA5 ISA 7.12.2, wave32, f16 16x16x32):
//  A (16x32):  row m = lane&15 ; g = lane>>4
//     halves e0..e7  <-> k = g*8 .. g*8+7      (contiguous!)
//     halves e8..e15 <-> k = 16+g*8 .. 16+g*8+7 (contiguous!)
//  B (32x16):  col n = lane&15 ; same k mapping (mirrored role)
//  C/D f32 16x16: element r (0..7): m = r + 8*g ; n = lane&15
// ---------------------------------------------------------------------------
__device__ __forceinline__ int frag_k(int g, int e) {
  int v = e >> 1, lo = e & 1;
  return (v < 4) ? (g * 8 + v * 2 + lo) : (16 + g * 8 + (v - 4) * 2 + lo);
}

__device__ __forceinline__ v16h frag_cat(v8h lo, v8h hi) {
  return __builtin_shufflevector(lo, hi, 0, 1, 2, 3, 4, 5, 6, 7,
                                         8, 9, 10, 11, 12, 13, 14, 15);
}

// load a fragment whose K runs contiguously at `base` (halves), 16B-aligned
__device__ __forceinline__ v16h frag_ldk(const _Float16* base, int g) {
  v8h lo = *(const v8h*)(base + g * 8);
  v8h hi = *(const v8h*)(base + 16 + g * 8);
  return frag_cat(lo, hi);
}

__device__ __forceinline__ void cvt16_store(float4 f0, float4 f1, float4 f2,
                                            float4 f3, _Float16* dst) {
  v8h h0, h1;
  h0[0] = (_Float16)f0.x; h0[1] = (_Float16)f0.y;
  h0[2] = (_Float16)f0.z; h0[3] = (_Float16)f0.w;
  h0[4] = (_Float16)f1.x; h0[5] = (_Float16)f1.y;
  h0[6] = (_Float16)f1.z; h0[7] = (_Float16)f1.w;
  h1[0] = (_Float16)f2.x; h1[1] = (_Float16)f2.y;
  h1[2] = (_Float16)f2.z; h1[3] = (_Float16)f2.w;
  h1[4] = (_Float16)f3.x; h1[5] = (_Float16)f3.y;
  h1[6] = (_Float16)f3.z; h1[7] = (_Float16)f3.w;
  *(v8h*)dst       = h0;
  *(v8h*)(dst + 8) = h1;
}

__device__ __forceinline__ float gelu_tanh(float x) {
  float x3 = x * x * x;
  return 0.5f * x * (1.0f + tanhf(0.7978845608028654f * (x + 0.044715f * x3)));
}

// ---------------------------------------------------------------------------
// Embedding + sinusoidal positional encoding -> f32 x[BS, D]
// ---------------------------------------------------------------------------
__global__ __launch_bounds__(256) void embed_pos_k(
    const int* __restrict__ tok, const float* __restrict__ emb,
    float* __restrict__ X)
{
  const int row = blockIdx.x, t = threadIdx.x;
  const int s = row % S_LEN;
  const int token = tok[row];
  const float sf = -9.210340371976184f / 255.0f;  // -ln(1e4)/(D/2-1)
  #pragma unroll
  for (int half = 0; half < 2; ++half) {
    int d = t + half * 256;
    float freq = __expf((float)(d & 255) * sf);
    float ang = (float)s * freq;
    float pe = (d < 256) ? __sinf(ang) : __cosf(ang);
    X[(long)row * D_DIM + d] = emb[(long)token * D_DIM + d] + pe;
  }
}

// ---------------------------------------------------------------------------
// Row LayerNorm over D=512; one block per row; OUT16: f16 (GEMM feed) else f32
// ---------------------------------------------------------------------------
template<bool OUT16>
__global__ __launch_bounds__(256) void layernorm_k(
    const float* __restrict__ X, const float* __restrict__ sc,
    const float* __restrict__ bi, _Float16* __restrict__ y16,
    float* __restrict__ y32)
{
  __shared__ float bs[256], bq[256];
  const int row = blockIdx.x, t = threadIdx.x;
  float v0 = X[(long)row * D_DIM + t];
  float v1 = X[(long)row * D_DIM + 256 + t];
  bs[t] = v0 + v1;
  bq[t] = v0 * v0 + v1 * v1;
  __syncthreads();
  for (int st = 128; st > 0; st >>= 1) {
    if (t < st) { bs[t] += bs[t + st]; bq[t] += bq[t + st]; }
    __syncthreads();
  }
  float mu   = bs[0] * (1.0f / (float)D_DIM);
  float var  = bq[0] * (1.0f / (float)D_DIM) - mu * mu;
  float rstd = rsqrtf(var + 1e-6f);
  float o0 = (v0 - mu) * rstd * sc[t]       + bi[t];
  float o1 = (v1 - mu) * rstd * sc[256 + t] + bi[256 + t];
  if (OUT16) {
    y16[(long)row * D_DIM + t]       = (_Float16)o0;
    y16[(long)row * D_DIM + 256 + t] = (_Float16)o1;
  } else {
    y32[(long)row * D_DIM + t]       = o0;
    y32[(long)row * D_DIM + 256 + t] = o1;
  }
}

// ---------------------------------------------------------------------------
// WMMA GEMM: C[M,N] = A_f16[M,K] * B_f32[K,N]  (B converted to f16 in LDS)
// 128 thr (4 waves), 64x64 tile, K staged 32/step, double-buffered pipeline:
//   iter t: issue global loads for tile t+1 -> compute tile t from LDS ->
//           ds_store tile t+1 into alternate buffer -> one barrier.
// Wave w owns column strip [w*16,w*16+16) over all 64 rows (1 bF gather +
// 4 vectorized aF per k-step). Weight tile t+2 prefetched into GL2.
// ---------------------------------------------------------------------------
template<bool BIAS, bool GELU, bool ACCF32>
__global__ __launch_bounds__(128) void wmma_gemm(
    const _Float16* __restrict__ A, const float* __restrict__ Bw,
    const float* __restrict__ bias, _Float16* __restrict__ out16,
    float* __restrict__ out32, int K, int N, float scale)
{
  __shared__ _Float16 sA[2][64 * SA_STRIDE];
  __shared__ _Float16 sB[2][32 * SK_STRIDE];
  const int tid  = threadIdx.x;
  const int lane = tid & 31, wv = tid >> 5;
  const int nn   = lane & 15, g = lane >> 4;
  const int row0 = blockIdx.y * 64, col0 = blockIdx.x * 64;

  int kmap[16];
  #pragma unroll
  for (int e = 0; e < 16; ++e) kmap[e] = frag_k(g, e);

  v8f acc[4];  // 4 independent row-subtile accumulator chains
  #pragma unroll
  for (int i = 0; i < 4; ++i)
    #pragma unroll
    for (int r = 0; r < 8; ++r) acc[i][r] = 0.0f;

  const int rowA = tid >> 1, segA = tid & 1;     // A staging: 64 rows x 2 segs
  const int rowB = tid >> 2, segB = tid & 3;     // B staging: 32 rows x 4 segs
  const _Float16* gA = &A[(long)(row0 + rowA) * K + segA * 16];
  const float*    gB = &Bw[(long)rowB * N + col0 + segB * 16];
  _Float16* dstA0 = &sA[0][rowA * SA_STRIDE + segA * 16];
  _Float16* dstB0 = &sB[0][rowB * SK_STRIDE + segB * 16];
  const int bufA = 64 * SA_STRIDE, bufB = 32 * SK_STRIDE;

  // prologue: stage tile 0 into buffer 0
  {
    v8h a0 = *(const v8h*)gA, a1 = *(const v8h*)(gA + 8);
    float4 f0 = ((const float4*)gB)[0], f1 = ((const float4*)gB)[1];
    float4 f2 = ((const float4*)gB)[2], f3 = ((const float4*)gB)[3];
    *(v8h*)dstA0 = a0; *(v8h*)(dstA0 + 8) = a1;
    cvt16_store(f0, f1, f2, f3, dstB0);
  }
  __syncthreads();

  const int nsteps = K >> 5;
  for (int t = 0; t < nsteps; ++t) {
    const int p = t & 1;
    const bool more = (t + 1 < nsteps);
    v8h na0 = {}, na1 = {};
    float4 nf0 = {}, nf1 = {}, nf2 = {}, nf3 = {};
    if (more) {                                  // issue next tile's loads now
      const _Float16* ga = gA + (t + 1) * 32;
      na0 = *(const v8h*)ga; na1 = *(const v8h*)(ga + 8);
      const float* gb = gB + (long)(t + 1) * 32 * N;
      nf0 = ((const float4*)gb)[0]; nf1 = ((const float4*)gb)[1];
      nf2 = ((const float4*)gb)[2]; nf3 = ((const float4*)gb)[3];
      if (t + 2 < nsteps)                        // GL2 prefetch, 2 steps ahead
        __builtin_prefetch(gb + (long)32 * N, 0, 1);
    }

    // compute tile t from buffer p (loads above hide behind these WMMAs)
    v16h bF;
    #pragma unroll
    for (int e = 0; e < 16; ++e)
      bF[e] = sB[p][kmap[e] * SK_STRIDE + wv * 16 + nn];
    #pragma unroll
    for (int i = 0; i < 4; ++i) {
      v16h aF = frag_ldk(&sA[p][(i * 16 + nn) * SA_STRIDE], g);
      acc[i] = __builtin_amdgcn_wmma_f32_16x16x32_f16(
                 false, aF, false, bF, (short)0, acc[i], false, false);
    }

    if (more) {                                  // drain loads into buffer p^1
      _Float16* da = dstA0 + (p ^ 1) * bufA;
      *(v8h*)da = na0; *(v8h*)(da + 8) = na1;
      cvt16_store(nf0, nf1, nf2, nf3, dstB0 + (p ^ 1) * bufB);
    }
    __syncthreads();
  }

  #pragma unroll
  for (int i = 0; i < 4; ++i) {
    #pragma unroll
    for (int r = 0; r < 8; ++r) {
      int row = row0 + i * 16 + r + 8 * g;
      int col = col0 + wv * 16 + nn;
      float v = acc[i][r];
      if (BIAS) v += bias[col];
      if (GELU) v = gelu_tanh(v);
      if (ACCF32) out32[(long)row * N + col] += v;
      else        out16[(long)row * N + col] = (_Float16)(v * scale);
    }
  }
}

// ---------------------------------------------------------------------------
// Sliding-window attention, flash-style online softmax.
// Block = 128 thr (4 waves) covering 64 consecutive queries of one (b,h).
// K/V chunks (32 keys) of the shared band staged once per block into
// double-buffered LDS; next chunk's global loads issue before this chunk's
// WMMAs and drain to LDS after them (2 barriers/chunk). Logits use two
// independent WMMA accumulators per tile to avoid RAW hazard stalls.
// ---------------------------------------------------------------------------
__global__ __launch_bounds__(128) void attn_window(
    const _Float16* __restrict__ Q, const _Float16* __restrict__ Km,
    const _Float16* __restrict__ Vm, const int* __restrict__ tok,
    _Float16* __restrict__ O16)
{
  const int h = blockIdx.y, b = blockIdx.z;
  const int tid  = threadIdx.x;
  const int lane = tid & 31, wv = tid >> 5;
  const int nn   = lane & 15, g = lane >> 4;

  __shared__ _Float16 sK[2][32 * SK_STRIDE];
  __shared__ _Float16 sV[2][32 * SK_STRIDE];
  __shared__ _Float16 sP[4][16 * SA_STRIDE];

  int kmap[16];
  #pragma unroll
  for (int e = 0; e < 16; ++e) kmap[e] = frag_k(g, e);

  const int qblk0 = blockIdx.x * 64;        // block's first query
  const int q0    = qblk0 + wv * 16;        // this wave's query tile

  const long rowQ = ((long)b * S_LEN + q0 + nn) * D_DIM + h * DH_DIM;
  v16h qA0 = frag_ldk(&Q[rowQ], g);
  v16h qA1 = frag_ldk(&Q[rowQ + 32], g);

  bool padq[8];
  #pragma unroll
  for (int r = 0; r < 8; ++r) padq[r] = tok[b * S_LEN + q0 + r + 8 * g] > 0;

  float mrow[8], lrow[8];
  v8f Oacc[4];
  #pragma unroll
  for (int r = 0; r < 8; ++r) { mrow[r] = -3.0e38f; lrow[r] = 0.0f; }
  #pragma unroll
  for (int j = 0; j < 4; ++j)
    #pragma unroll
    for (int r = 0; r < 8; ++r) Oacc[j][r] = 0.0f;

  // block-uniform key band (union of the 4 waves' bands)
  int lo = qblk0 - WIN_HALF; if (lo < 0) lo = 0; lo &= ~31;
  int hi = qblk0 + 63 + WIN_HALF + 1; if (hi > S_LEN) hi = S_LEN;
  hi = (hi + 31) & ~31;

  const int srow = tid >> 2, sseg = tid & 3;   // staging: 32 rows x 4 segs
  const long gKV0 = ((long)b * S_LEN + srow) * D_DIM + h * DH_DIM + sseg * 16;
  _Float16* dK0 = &sK[0][srow * SK_STRIDE + sseg * 16];
  _Float16* dV0 = &sV[0][srow * SK_STRIDE + sseg * 16];
  const int bufKV = 32 * SK_STRIDE;

  // prologue: stage first chunk into buffer 0
  {
    const long gk = gKV0 + (long)lo * D_DIM;
    v8h ka = *(const v8h*)&Km[gk], kb_ = *(const v8h*)&Km[gk + 8];
    v8h va = *(const v8h*)&Vm[gk], vb_ = *(const v8h*)&Vm[gk + 8];
    *(v8h*)dK0 = ka; *(v8h*)(dK0 + 8) = kb_;
    *(v8h*)dV0 = va; *(v8h*)(dV0 + 8) = vb_;
  }
  __syncthreads();

  const int nch = (hi - lo) >> 5;
  for (int t = 0; t < nch; ++t) {
    const int p  = t & 1;
    const int kb = lo + t * 32;
    const bool more = (t + 1 < nch);
    v8h nka = {}, nkb = {}, nva = {}, nvb = {};
    if (more) {                               // issue next chunk's loads now
      const long gk = gKV0 + (long)(kb + 32) * D_DIM;
      nka = *(const v8h*)&Km[gk]; nkb = *(const v8h*)&Km[gk + 8];
      nva = *(const v8h*)&Vm[gk]; nvb = *(const v8h*)&Vm[gk + 8];
    }

    // logits: S = q . k^T; two independent accumulators per 16-key tile
    const _Float16* kr0 = &sK[p][nn * SK_STRIDE];
    const _Float16* kr1 = &sK[p][(16 + nn) * SK_STRIDE];
    v8f z = {0.f, 0.f, 0.f, 0.f, 0.f, 0.f, 0.f, 0.f};
    v8f Sa0 = __builtin_amdgcn_wmma_f32_16x16x32_f16(
                false, qA0, false, frag_ldk(kr0, g),      (short)0, z, false, false);
    v8f Sb0 = __builtin_amdgcn_wmma_f32_16x16x32_f16(
                false, qA1, false, frag_ldk(kr0 + 32, g), (short)0, z, false, false);
    v8f Sa1 = __builtin_amdgcn_wmma_f32_16x16x32_f16(
                false, qA0, false, frag_ldk(kr1, g),      (short)0, z, false, false);
    v8f Sb1 = __builtin_amdgcn_wmma_f32_16x16x32_f16(
                false, qA1, false, frag_ldk(kr1 + 32, g), (short)0, z, false, false);
    v8f Sc0 = Sa0 + Sb0;
    v8f Sc1 = Sa1 + Sb1;

    // band + padding mask
    const bool pk0 = tok[b * S_LEN + kb + nn] > 0;
    const bool pk1 = tok[b * S_LEN + kb + 16 + nn] > 0;
    float chm[8];
    #pragma unroll
    for (int r = 0; r < 8; ++r) {
      int qi = q0 + r + 8 * g;
      int d0 = qi - (kb + nn);      if (d0 < 0) d0 = -d0;
      int d1 = qi - (kb + 16 + nn); if (d1 < 0) d1 = -d1;
      float s0 = Sc0[r] + ((d0 <= WIN_HALF && padq[r] && pk0) ? 0.0f : NEG_BIAS);
      float s1 = Sc1[r] + ((d1 <= WIN_HALF && padq[r] && pk1) ? 0.0f : NEG_BIAS);
      Sc0[r] = s0; Sc1[r] = s1;
      chm[r] = fmaxf(s0, s1);
    }
    #pragma unroll
    for (int off = 1; off < 16; off <<= 1)
      #pragma unroll
      for (int r = 0; r < 8; ++r)
        chm[r] = fmaxf(chm[r], __shfl_xor(chm[r], off, 16));

    // online softmax update
    float rs[8];
    #pragma unroll
    for (int r = 0; r < 8; ++r) {
      float mn = fmaxf(mrow[r], chm[r]);
      float alpha = __expf(mrow[r] - mn);
      mrow[r] = mn;
      float p0 = __expf(Sc0[r] - mn);
      float p1 = __expf(Sc1[r] - mn);
      Sc0[r] = p0; Sc1[r] = p1;
      rs[r] = p0 + p1;
      lrow[r] *= alpha;
      #pragma unroll
      for (int j = 0; j < 4; ++j) Oacc[j][r] *= alpha;
    }
    #pragma unroll
    for (int off = 1; off < 16; off <<= 1)
      #pragma unroll
      for (int r = 0; r < 8; ++r) rs[r] += __shfl_xor(rs[r], off, 16);
    #pragma unroll
    for (int r = 0; r < 8; ++r) lrow[r] += rs[r];

    // re-layout P (C-fragment) -> A-fragment via this wave's LDS slice
    #pragma unroll
    for (int r = 0; r < 8; ++r) {
      sP[wv][(r + 8 * g) * SA_STRIDE + nn]      = (_Float16)Sc0[r];
      sP[wv][(r + 8 * g) * SA_STRIDE + 16 + nn] = (_Float16)Sc1[r];
    }
    __syncthreads();                       // block-uniform trip count -> safe
    v16h pF = frag_ldk(&sP[wv][nn * SA_STRIDE], g);

    // O += P . V  (4 independent accumulator chains)
    #pragma unroll
    for (int j = 0; j < 4; ++j) {
      v16h vB;
      #pragma unroll
      for (int e = 0; e < 16; ++e)
        vB[e] = sV[p][kmap[e] * SK_STRIDE + j * 16 + nn];
      Oacc[j] = __builtin_amdgcn_wmma_f32_16x16x32_f16(
                  false, pF, false, vB, (short)0, Oacc[j], false, false);
    }

    if (more) {                            // drain next chunk into buffer p^1
      _Float16* dk = dK0 + (p ^ 1) * bufKV;
      _Float16* dv = dV0 + (p ^ 1) * bufKV;
      *(v8h*)dk = nka; *(v8h*)(dk + 8) = nkb;
      *(v8h*)dv = nva; *(v8h*)(dv + 8) = nvb;
    }
    __syncthreads();
  }

  float inv[8];
  #pragma unroll
  for (int r = 0; r < 8; ++r) inv[r] = (lrow[r] > 0.0f) ? (1.0f / lrow[r]) : 0.0f;
  #pragma unroll
  for (int j = 0; j < 4; ++j)
    #pragma unroll
    for (int r = 0; r < 8; ++r)
      O16[((long)b * S_LEN + q0 + r + 8 * g) * D_DIM + h * DH_DIM + j * 16 + nn] =
          (_Float16)(Oacc[j][r] * inv[r]);
}

// ---------------------------------------------------------------------------
// Host-side orchestration
// ---------------------------------------------------------------------------
extern "C" void kernel_launch(void* const* d_in, const int* in_sizes, int n_in,
                              void* d_out, int out_size, void* d_ws, size_t ws_size,
                              hipStream_t stream) {
  (void)in_sizes; (void)n_in; (void)out_size; (void)ws_size;
  const int*   tok  = (const int*)  d_in[0];
  const float* emb  = (const float*)d_in[1];
  const float* wq   = (const float*)d_in[2];
  const float* wk   = (const float*)d_in[3];
  const float* wv   = (const float*)d_in[4];
  const float* wo   = (const float*)d_in[5];
  const float* ln1s = (const float*)d_in[6];
  const float* ln1b = (const float*)d_in[7];
  const float* ln2s = (const float*)d_in[8];
  const float* ln2b = (const float*)d_in[9];
  const float* w1   = (const float*)d_in[10];
  const float* b1   = (const float*)d_in[11];
  const float* w2   = (const float*)d_in[12];
  const float* b2   = (const float*)d_in[13];
  const float* lnfs = (const float*)d_in[14];
  const float* lnfb = (const float*)d_in[15];

  char* ws = (char*)d_ws;
  const int BS = 2 * S_LEN;  // 4096 rows
  float*    X  = (float*)   (ws);                          //  8 MB f32 residual
  _Float16* Y  = (_Float16*)(ws + 8ull  * 1024 * 1024);    //  4 MB LN out (f16)
  _Float16* Qb = (_Float16*)(ws + 12ull * 1024 * 1024);    //  4 MB
  _Float16* Kb = (_Float16*)(ws + 16ull * 1024 * 1024);    //  4 MB
  _Float16* Vb = (_Float16*)(ws + 20ull * 1024 * 1024);    //  4 MB
  _Float16* Ob = (_Float16*)(ws + 24ull * 1024 * 1024);    //  4 MB attn out
  _Float16* Hb = (_Float16*)(ws + 28ull * 1024 * 1024);    // 16 MB MLP hidden

  embed_pos_k<<<BS, 256, 0, stream>>>(tok, emb, X);

  const float qscale = 0.125f;  // 1/sqrt(DH)
  dim3 gN512(512 / 64, BS / 64), gN2048(2048 / 64, BS / 64);
  dim3 gAttn(S_LEN / 64, H_NUM, 2);

  for (int l = 0; l < L_NUM; ++l) {
    layernorm_k<true><<<BS, 256, 0, stream>>>(X, ln1s + (size_t)l * D_DIM,
                                              ln1b + (size_t)l * D_DIM, Y, nullptr);
    wmma_gemm<false, false, false><<<gN512, 128, 0, stream>>>(
        Y, wq + (size_t)l * D_DIM * D_DIM, nullptr, Qb, nullptr, 512, 512, qscale);
    wmma_gemm<false, false, false><<<gN512, 128, 0, stream>>>(
        Y, wk + (size_t)l * D_DIM * D_DIM, nullptr, Kb, nullptr, 512, 512, 1.0f);
    wmma_gemm<false, false, false><<<gN512, 128, 0, stream>>>(
        Y, wv + (size_t)l * D_DIM * D_DIM, nullptr, Vb, nullptr, 512, 512, 1.0f);
    attn_window<<<gAttn, 128, 0, stream>>>(Qb, Kb, Vb, tok, Ob);
    wmma_gemm<false, false, true><<<gN512, 128, 0, stream>>>(
        Ob, wo + (size_t)l * D_DIM * D_DIM, nullptr, nullptr, X, 512, 512, 1.0f);
    layernorm_k<true><<<BS, 256, 0, stream>>>(X, ln2s + (size_t)l * D_DIM,
                                              ln2b + (size_t)l * D_DIM, Y, nullptr);
    wmma_gemm<true, true, false><<<gN2048, 128, 0, stream>>>(
        Y, w1 + (size_t)l * D_DIM * M_DIM, b1 + (size_t)l * M_DIM, Hb, nullptr,
        512, 2048, 1.0f);
    wmma_gemm<true, false, true><<<gN512, 128, 0, stream>>>(
        Hb, w2 + (size_t)l * M_DIM * D_DIM, b2 + (size_t)l * D_DIM, nullptr, X,
        2048, 512, 1.0f);
  }
  layernorm_k<false><<<BS, 256, 0, stream>>>(X, lnfs, lnfb, nullptr, (float*)d_out);
}
